// CoBFormer_74251394614028
// MI455X (gfx1250) — matrix operand order repaired and checked by reading, not compiled, not run on hardware
//
#include <hip/hip_runtime.h>

typedef __bf16 bf16;
typedef __attribute__((ext_vector_type(16))) __bf16 v16bf;
typedef __attribute__((ext_vector_type(8)))  __bf16 v8bf;
typedef __attribute__((ext_vector_type(8)))  float  v8f;

#if defined(__has_builtin)
#  if __has_builtin(__builtin_amdgcn_global_load_async_to_lds_b128) && \
      __has_builtin(__builtin_amdgcn_s_wait_asynccnt)
#    define USE_ASYNC_LDS 1
#  endif
#endif
#ifndef USE_ASYNC_LDS
#  define USE_ASYNC_LDS 0
#endif

#if USE_ASYNC_LDS
typedef int v4i_gcc __attribute__((vector_size(16)));
typedef __attribute__((address_space(1))) v4i_gcc* as1_i4p;
typedef __attribute__((address_space(3))) v4i_gcc* as3_i4p;
#endif

#define CAT16(lo, hi) __builtin_shufflevector((lo), (hi), 0,1,2,3,4,5,6,7,8,9,10,11,12,13,14,15)

__device__ __forceinline__ v8f fzero8() {
  v8f z;
#pragma unroll
  for (int i = 0; i < 8; ++i) z[i] = 0.0f;
  return z;
}
__device__ __forceinline__ v8bf bzero8() {
  v8bf z;
#pragma unroll
  for (int i = 0; i < 8; ++i) z[i] = (bf16)0.0f;
  return z;
}

// A-fragment (16x32, 16-bit): lane m = lane%16; lanes<16 hold K {0..7,16..23}, lanes>=16 hold K {8..15,24..31}
template <int PITCH>
__device__ __forceinline__ v16bf frag_a(const bf16 (*tile)[PITCH], int mrow, int kbase, int lane) {
  const bf16* p = &tile[mrow][kbase + ((lane < 16) ? 0 : 8)];
  v8bf lo = *(const v8bf*)p;
  v8bf hi = *(const v8bf*)(p + 16);
  return CAT16(lo, hi);
}
// B-fragment (32x16, 16-bit) from transposed tile Bt[n][k]: lane n = lane%16; lanes<16 hold K {0..7,16..23}, lanes>=16 hold K {8..15,24..31}
template <int PITCH>
__device__ __forceinline__ v16bf frag_b(const bf16 (*tile)[PITCH], int ncol, int kbase, int lane) {
  const bf16* p = &tile[ncol][kbase + ((lane < 16) ? 0 : 8)];
  v8bf lo = *(const v8bf*)p;
  v8bf hi = *(const v8bf*)(p + 16);
  return CAT16(lo, hi);
}

// ---------------------------------------------------------------------------
// Generic bf16 GEMM: out = epilogue(A[M,K] @ Bt[N,K]^T)
//  - A2 (optional): for K=1024 fuse, k>=512 reads A2[(row>>6)*512 + k-512]
//  - bias fp32, relu (after bias), resid bf16 (after relu), outputs bf16/f32
//  - scatter (optional): remap output rows for bf16 output
//  - tiles staged to LDS with async global->LDS copies when available
// ---------------------------------------------------------------------------
__global__ __launch_bounds__(256) void gemm_bf16_kernel(
    const bf16* __restrict__ A, int lda,
    const bf16* __restrict__ A2,
    const bf16* __restrict__ Bt,
    const float* __restrict__ bias,
    const bf16* __restrict__ resid,
    bf16* __restrict__ outb, float* __restrict__ outf,
    const int* __restrict__ scatter,
    int M, int N, int K, int relu) {
  __shared__ bf16 sA[2][128][40];
  __shared__ bf16 sB[2][128][40];

  const int tid  = threadIdx.x;
  const int lane = tid & 31;
  const int wid  = tid >> 5;
  const int m0   = blockIdx.y * 128;
  const int n0   = blockIdx.x * 128;

  const int lrow = tid >> 1;        // 0..127
  const int lseg = (tid & 1) * 16;  // 0 or 16 (bf16 elems)
  const int nt   = K >> 5;

  v8f acc[2][4];
#pragma unroll
  for (int i = 0; i < 2; ++i)
#pragma unroll
    for (int j = 0; j < 4; ++j) acc[i][j] = fzero8();

  const int wm = wid & 3;   // 4 row groups of 32
  const int wn = wid >> 2;  // 2 col groups of 64
  const int mb = wm * 32;
  const int nb = wn * 64;
  const int l16 = lane & 15;

  auto srcA = [&](int kt) -> const bf16* {
    int gr = m0 + lrow;
    int kk = kt * 32 + lseg;
    if (A2 && kk >= 512) return A2 + (size_t)(gr >> 6) * 512 + (kk - 512);
    return A + (size_t)gr * lda + kk;
  };

#if USE_ASYNC_LDS
  // Async path: copy engine writes LDS directly; wave overlaps WMMA with copies.
  auto issueTile = [&](int kt, int buf) {
    bf16* dA = &sA[buf][lrow][lseg];
    bf16* dB = &sB[buf][lrow][lseg];
    if (m0 + lrow < M) {
      const bf16* src = srcA(kt);
      __builtin_amdgcn_global_load_async_to_lds_b128((as1_i4p)src, (as3_i4p)dA, 0, 0);
      __builtin_amdgcn_global_load_async_to_lds_b128((as1_i4p)(src + 8), (as3_i4p)(dA + 8), 0, 0);
    } else {
      *(v8bf*)dA = bzero8();
      *(v8bf*)(dA + 8) = bzero8();
    }
    if (n0 + lrow < N) {
      const bf16* src = Bt + (size_t)(n0 + lrow) * K + kt * 32 + lseg;
      __builtin_amdgcn_global_load_async_to_lds_b128((as1_i4p)src, (as3_i4p)dB, 0, 0);
      __builtin_amdgcn_global_load_async_to_lds_b128((as1_i4p)(src + 8), (as3_i4p)(dB + 8), 0, 0);
    } else {
      *(v8bf*)dB = bzero8();
      *(v8bf*)(dB + 8) = bzero8();
    }
  };

  issueTile(0, 0);
  __builtin_amdgcn_s_wait_asynccnt(0);
  __syncthreads();

  for (int t = 0; t < nt; ++t) {
    const int buf = t & 1;
    if (t + 1 < nt) issueTile(t + 1, buf ^ 1);
    v16bf af[2], bfv[4];
#pragma unroll
    for (int i = 0; i < 2; ++i) af[i] = frag_a<40>(sA[buf], mb + 16 * i + l16, 0, lane);
#pragma unroll
    for (int j = 0; j < 4; ++j) bfv[j] = frag_b<40>(sB[buf], nb + 16 * j + l16, 0, lane);
#pragma unroll
    for (int i = 0; i < 2; ++i)
#pragma unroll
      for (int j = 0; j < 4; ++j)
        acc[i][j] = __builtin_amdgcn_wmma_f32_16x16x32_bf16(
            false, af[i], false, bfv[j], (short)0, acc[i][j], false, false);
    if (t + 1 < nt) {
      __builtin_amdgcn_s_wait_asynccnt(0);
      __syncthreads();
    }
  }
#else
  // Fallback: stage through VGPRs with double buffering + global prefetch.
  v8bf ra[2], rb[2];
  auto loadA = [&](int kt) {
    if (m0 + lrow < M) {
      const bf16* src = srcA(kt);
      ra[0] = *(const v8bf*)src;
      ra[1] = *(const v8bf*)(src + 8);
    } else { ra[0] = bzero8(); ra[1] = bzero8(); }
  };
  auto loadB = [&](int kt) {
    int gn = n0 + lrow;
    int kk = kt * 32 + lseg;
    if (gn < N) {
      const bf16* src = Bt + (size_t)gn * K + kk;
      rb[0] = *(const v8bf*)src;
      rb[1] = *(const v8bf*)(src + 8);
    } else { rb[0] = bzero8(); rb[1] = bzero8(); }
  };
  auto stage = [&](int buf) {
    *(v8bf*)&sA[buf][lrow][lseg]     = ra[0];
    *(v8bf*)&sA[buf][lrow][lseg + 8] = ra[1];
    *(v8bf*)&sB[buf][lrow][lseg]     = rb[0];
    *(v8bf*)&sB[buf][lrow][lseg + 8] = rb[1];
  };

  loadA(0); loadB(0);
  stage(0);
  __syncthreads();

  for (int t = 0; t < nt; ++t) {
    const int buf = t & 1;
    if (t + 1 < nt) { loadA(t + 1); loadB(t + 1); }
    if (t + 2 < nt) {
      int kk = (t + 2) * 32 + lseg;
      if ((m0 + lrow < M) && (!A2 || kk < 512))
        __builtin_prefetch(A + (size_t)(m0 + lrow) * lda + kk, 0, 3);
      if (n0 + lrow < N)
        __builtin_prefetch(Bt + (size_t)(n0 + lrow) * K + kk, 0, 3);
    }
    v16bf af[2], bfv[4];
#pragma unroll
    for (int i = 0; i < 2; ++i) af[i] = frag_a<40>(sA[buf], mb + 16 * i + l16, 0, lane);
#pragma unroll
    for (int j = 0; j < 4; ++j) bfv[j] = frag_b<40>(sB[buf], nb + 16 * j + l16, 0, lane);
#pragma unroll
    for (int i = 0; i < 2; ++i)
#pragma unroll
      for (int j = 0; j < 4; ++j)
        acc[i][j] = __builtin_amdgcn_wmma_f32_16x16x32_bf16(
            false, af[i], false, bfv[j], (short)0, acc[i][j], false, false);
    if (t + 1 < nt) {
      stage((t + 1) & 1);
      __syncthreads();
    }
  }
#endif

  // Epilogue: C/D layout — lane n = lane%16; VGPR r: M = r (lanes<16) or r+8 (lanes>=16)
  const int radd = (lane < 16) ? 0 : 8;
#pragma unroll
  for (int j = 0; j < 4; ++j) {
    int col = n0 + nb + 16 * j + l16;
    if (col >= N) continue;
    float bv = bias ? bias[col] : 0.0f;
#pragma unroll
    for (int i = 0; i < 2; ++i) {
#pragma unroll
      for (int r = 0; r < 8; ++r) {
        int row = m0 + mb + 16 * i + r + radd;
        if (row >= M) continue;
        float v = acc[i][j][r] + bv;
        if (relu) v = fmaxf(v, 0.0f);
        if (resid) v += (float)resid[(size_t)row * N + col];
        if (outb) {
          int orow = scatter ? scatter[row] : row;
          outb[(size_t)orow * N + col] = (bf16)v;
        }
        if (outf) outf[(size_t)row * N + col] = v;
      }
    }
  }
}

// ---------------------------------------------------------------------------
// Per-patch node attention: one block per (patch, head); 64x64 tiles, d=64.
// o may alias q (q is fully staged to LDS before writes).
// ---------------------------------------------------------------------------
__global__ __launch_bounds__(128) void node_attn_kernel(
    const bf16* __restrict__ q, const bf16* __restrict__ k,
    const bf16* __restrict__ v, bf16* __restrict__ o,
    const int* __restrict__ patch, int C, int padNode) {
  __shared__ bf16 sQ[64][72];
  __shared__ bf16 sK[64][72];
  __shared__ bf16 sVt[64][72];
  __shared__ bf16 sP[64][72];
  __shared__ float sS[64][68];
  __shared__ int sIdx[64];

  const int p    = blockIdx.x;
  const int head = blockIdx.y;
  const int tid  = threadIdx.x;
  const int lane = tid & 31;
  const int wid  = tid >> 5;
  const size_t rbase = (size_t)p * 64;
  const int cbase = head * 64;

  if (tid < 64) sIdx[tid] = patch[p * 64 + tid];
  for (int idx = tid; idx < 4096; idx += 128) {
    int s = idx >> 6, d = idx & 63;
    size_t go = (rbase + s) * C + cbase + d;
    sQ[s][d]  = (bf16)((float)q[go] * 0.125f);  // 1/sqrt(64)
    sK[s][d]  = k[go];
    sVt[d][s] = v[go];
  }
  __syncthreads();

  const int wm = wid & 1, wn = wid >> 1;
  const int mb = wm * 32, nb = wn * 32;
  const int l16 = lane & 15;
  const int radd = (lane < 16) ? 0 : 8;

  v8f sc[2][2];
#pragma unroll
  for (int i = 0; i < 2; ++i)
#pragma unroll
    for (int j = 0; j < 2; ++j) sc[i][j] = fzero8();

#pragma unroll
  for (int kt = 0; kt < 2; ++kt) {
    v16bf af[2], bfv[2];
#pragma unroll
    for (int i = 0; i < 2; ++i) af[i] = frag_a<72>(sQ, mb + 16 * i + l16, kt * 32, lane);
#pragma unroll
    for (int j = 0; j < 2; ++j) bfv[j] = frag_b<72>(sK, nb + 16 * j + l16, kt * 32, lane);
#pragma unroll
    for (int i = 0; i < 2; ++i)
#pragma unroll
      for (int j = 0; j < 2; ++j)
        sc[i][j] = __builtin_amdgcn_wmma_f32_16x16x32_bf16(
            false, af[i], false, bfv[j], (short)0, sc[i][j], false, false);
  }
#pragma unroll
  for (int i = 0; i < 2; ++i)
#pragma unroll
    for (int j = 0; j < 2; ++j)
#pragma unroll
      for (int r = 0; r < 8; ++r) {
        int row = mb + 16 * i + r + radd;
        int col = nb + 16 * j + l16;
        bool ok = (sIdx[row] != padNode) && (sIdx[col] != padNode);
        sS[row][col] = ok ? sc[i][j][r] : -1e9f;
      }
  __syncthreads();

  if (tid < 64) {
    float m = -3.4e38f;
    for (int c = 0; c < 64; ++c) m = fmaxf(m, sS[tid][c]);
    float sum = 0.0f;
    for (int c = 0; c < 64; ++c) { float e = __expf(sS[tid][c] - m); sum += e; sS[tid][c] = e; }
    float inv = 1.0f / sum;
    for (int c = 0; c < 64; ++c) sP[tid][c] = (bf16)(sS[tid][c] * inv);
  }
  __syncthreads();

  v8f oc[2][2];
#pragma unroll
  for (int i = 0; i < 2; ++i)
#pragma unroll
    for (int j = 0; j < 2; ++j) oc[i][j] = fzero8();
#pragma unroll
  for (int kt = 0; kt < 2; ++kt) {
    v16bf af[2], bfv[2];
#pragma unroll
    for (int i = 0; i < 2; ++i) af[i] = frag_a<72>(sP, mb + 16 * i + l16, kt * 32, lane);
#pragma unroll
    for (int j = 0; j < 2; ++j) bfv[j] = frag_b<72>(sVt, nb + 16 * j + l16, kt * 32, lane);
#pragma unroll
    for (int i = 0; i < 2; ++i)
#pragma unroll
      for (int j = 0; j < 2; ++j)
        oc[i][j] = __builtin_amdgcn_wmma_f32_16x16x32_bf16(
            false, af[i], false, bfv[j], (short)0, oc[i][j], false, false);
  }
#pragma unroll
  for (int i = 0; i < 2; ++i)
#pragma unroll
    for (int j = 0; j < 2; ++j)
#pragma unroll
      for (int r = 0; r < 8; ++r) {
        int row = mb + 16 * i + r + radd;
        int col = nb + 16 * j + l16;
        o[(rbase + row) * C + cbase + col] = (bf16)oc[i][j][r];
      }
}

// ---------------------------------------------------------------------------
// Patch-level flash attention: block per (q-tile of 64, head); online softmax
// over 64-wide key tiles. o may alias q.
// ---------------------------------------------------------------------------
__global__ __launch_bounds__(128) void patch_attn_kernel(
    const bf16* __restrict__ q, const bf16* __restrict__ k,
    const bf16* __restrict__ v, bf16* __restrict__ o, int Pn, int C) {
  __shared__ bf16 sQ[64][72];
  __shared__ bf16 sK[64][72];
  __shared__ bf16 sVt[64][72];
  __shared__ bf16 sP[64][72];
  __shared__ float sS[64][68];
  __shared__ float rowM[64], rowL[64], rowA[64];

  const int q0   = blockIdx.x * 64;
  const int head = blockIdx.y;
  const int tid  = threadIdx.x;
  const int lane = tid & 31;
  const int wid  = tid >> 5;
  const int cbase = head * 64;

  for (int idx = tid; idx < 4096; idx += 128) {
    int s = idx >> 6, d = idx & 63;
    int gr = q0 + s;
    sQ[s][d] = (gr < Pn) ? (bf16)((float)q[(size_t)gr * C + cbase + d] * 0.125f) : (bf16)0.0f;
  }
  if (tid < 64) { rowM[tid] = -3.4e38f; rowL[tid] = 0.0f; }

  const int wm = wid & 1, wn = wid >> 1;
  const int mb = wm * 32, nb = wn * 32;
  const int l16 = lane & 15;
  const int radd = (lane < 16) ? 0 : 8;

  v8f acc[2][2];
#pragma unroll
  for (int i = 0; i < 2; ++i)
#pragma unroll
    for (int j = 0; j < 2; ++j) acc[i][j] = fzero8();

  const int ntiles = (Pn + 63) >> 6;
  for (int kt2 = 0; kt2 < ntiles; ++kt2) {
    int k0 = kt2 * 64;
    for (int idx = tid; idx < 4096; idx += 128) {
      int s = idx >> 6, d = idx & 63;
      int gr = k0 + s;
      bf16 kv = (gr < Pn) ? k[(size_t)gr * C + cbase + d] : (bf16)0.0f;
      bf16 vv = (gr < Pn) ? v[(size_t)gr * C + cbase + d] : (bf16)0.0f;
      sK[s][d] = kv;
      sVt[d][s] = vv;
    }
    __syncthreads();

    v8f sc[2][2];
#pragma unroll
    for (int i = 0; i < 2; ++i)
#pragma unroll
      for (int j = 0; j < 2; ++j) sc[i][j] = fzero8();
#pragma unroll
    for (int kt = 0; kt < 2; ++kt) {
      v16bf af[2], bfv[2];
#pragma unroll
      for (int i = 0; i < 2; ++i) af[i] = frag_a<72>(sQ, mb + 16 * i + l16, kt * 32, lane);
#pragma unroll
      for (int j = 0; j < 2; ++j) bfv[j] = frag_b<72>(sK, nb + 16 * j + l16, kt * 32, lane);
#pragma unroll
      for (int i = 0; i < 2; ++i)
#pragma unroll
        for (int j = 0; j < 2; ++j)
          sc[i][j] = __builtin_amdgcn_wmma_f32_16x16x32_bf16(
              false, af[i], false, bfv[j], (short)0, sc[i][j], false, false);
    }
#pragma unroll
    for (int i = 0; i < 2; ++i)
#pragma unroll
      for (int j = 0; j < 2; ++j)
#pragma unroll
        for (int r = 0; r < 8; ++r) {
          int row = mb + 16 * i + r + radd;
          int col = nb + 16 * j + l16;
          sS[row][col] = (k0 + col < Pn) ? sc[i][j][r] : -1e9f;
        }
    __syncthreads();

    if (tid < 64) {
      float m = rowM[tid];
      float tm = -3.4e38f;
      for (int c = 0; c < 64; ++c) tm = fmaxf(tm, sS[tid][c]);
      float nm = fmaxf(m, tm);
      float alpha = __expf(m - nm);
      float sum = 0.0f;
      for (int c = 0; c < 64; ++c) {
        float e = __expf(sS[tid][c] - nm);
        sum += e;
        sP[tid][c] = (bf16)e;
      }
      rowA[tid] = alpha;
      rowM[tid] = nm;
      rowL[tid] = rowL[tid] * alpha + sum;
    }
    __syncthreads();

#pragma unroll
    for (int i = 0; i < 2; ++i)
#pragma unroll
      for (int j = 0; j < 2; ++j)
#pragma unroll
        for (int r = 0; r < 8; ++r)
          acc[i][j][r] *= rowA[mb + 16 * i + r + radd];
#pragma unroll
    for (int kt = 0; kt < 2; ++kt) {
      v16bf af[2], bfv[2];
#pragma unroll
      for (int i = 0; i < 2; ++i) af[i] = frag_a<72>(sP, mb + 16 * i + l16, kt * 32, lane);
#pragma unroll
      for (int j = 0; j < 2; ++j) bfv[j] = frag_b<72>(sVt, nb + 16 * j + l16, kt * 32, lane);
#pragma unroll
      for (int i = 0; i < 2; ++i)
#pragma unroll
        for (int j = 0; j < 2; ++j)
          acc[i][j] = __builtin_amdgcn_wmma_f32_16x16x32_bf16(
              false, af[i], false, bfv[j], (short)0, acc[i][j], false, false);
    }
    __syncthreads();
  }

#pragma unroll
  for (int i = 0; i < 2; ++i)
#pragma unroll
    for (int j = 0; j < 2; ++j)
#pragma unroll
      for (int r = 0; r < 8; ++r) {
        int row = mb + 16 * i + r + radd;
        int gr = q0 + row;
        if (gr < Pn) {
          int col = nb + 16 * j + l16;
          float val = acc[i][j][r] / rowL[row];
          o[(size_t)gr * C + cbase + col] = (bf16)val;
        }
      }
}

// ---------------------------------------------------------------------------
// LayerNorm over C=512: one wave per row, shfl reduction.
// ---------------------------------------------------------------------------
__global__ __launch_bounds__(256) void ln_kernel(
    const bf16* __restrict__ x, bf16* __restrict__ y,
    const float* __restrict__ g, const float* __restrict__ b, int M, float eps) {
  const int C = 512;
  int row = blockIdx.x * 8 + (threadIdx.x >> 5);
  int lane = threadIdx.x & 31;
  if (row >= M) return;
  const bf16* xr = x + (size_t)row * C;
  float s = 0.0f, s2 = 0.0f;
  for (int c = lane; c < C; c += 32) {
    float v = (float)xr[c];
    s += v;
    s2 += v * v;
  }
#pragma unroll
  for (int o = 16; o > 0; o >>= 1) {
    s += __shfl_xor(s, o, 32);
    s2 += __shfl_xor(s2, o, 32);
  }
  float mu = s * (1.0f / C);
  float var = s2 * (1.0f / C) - mu * mu;
  float rs = rsqrtf(var + eps);
  bf16* yr = y + (size_t)row * C;
  for (int c = lane; c < C; c += 32) {
    float v = ((float)xr[c] - mu) * rs * g[c] + b[c];
    yr[c] = (bf16)v;
  }
}

// Gather: px[r] = hn[patch[r]]; one block per row, uint-wide copies.
__global__ __launch_bounds__(256) void gather_kernel(
    const bf16* __restrict__ src, const int* __restrict__ idx,
    bf16* __restrict__ dst, int C) {
  int r = blockIdx.x;
  int node = idx[r];
  const unsigned int* s = (const unsigned int*)(src + (size_t)node * C);
  unsigned int* d = (unsigned int*)(dst + (size_t)r * C);
  d[threadIdx.x] = s[threadIdx.x];  // C/2 == 256 uints
}

// Mean over S=64 per (patch, channel).
__global__ __launch_bounds__(256) void meanpool_kernel(
    const bf16* __restrict__ x, bf16* __restrict__ y, int Pn, int C) {
  int idx = blockIdx.x * 256 + threadIdx.x;
  if (idx >= Pn * C) return;
  int p = idx / C, c = idx - p * C;
  const bf16* row = x + ((size_t)p * 64) * C + c;
  float s = 0.0f;
  for (int j = 0; j < 64; ++j) s += (float)row[(size_t)j * C];
  y[idx] = (bf16)(s * (1.0f / 64.0f));
}

// Transpose+convert weights: Wt[n*K + k] = (bf16)W[k*N + n]
__global__ __launch_bounds__(256) void cvtT_kernel(
    const float* __restrict__ W, bf16* __restrict__ Wt, int K, int N) {
  int idx = blockIdx.x * 256 + threadIdx.x;
  if (idx >= K * N) return;
  int n = idx / K, kk = idx - n * K;
  Wt[idx] = (bf16)W[(size_t)kk * N + n];
}

// Elementwise fp32 -> bf16
__global__ __launch_bounds__(256) void cvt_kernel(
    const float* __restrict__ x, bf16* __restrict__ y, long long n) {
  long long i = (long long)blockIdx.x * 256 + threadIdx.x;
  if (i < n) y[i] = (bf16)x[i];
}

// ---------------------------------------------------------------------------
extern "C" void kernel_launch(void* const* d_in, const int* in_sizes, int n_in,
                              void* d_out, int out_size, void* d_ws, size_t ws_size,
                              hipStream_t stream) {
  (void)in_sizes; (void)n_in; (void)out_size; (void)ws_size;
  const float* X      = (const float*)d_in[0];
  const int*   patch  = (const int*)  d_in[1];
  const float* enc_w  = (const float*)d_in[2];
  const float* enc_b  = (const float*)d_in[3];
  const float* nn_g   = (const float*)d_in[4];
  const float* nn_b   = (const float*)d_in[5];
  const float* wq     = (const float*)d_in[6];
  const float* wk     = (const float*)d_in[7];
  const float* wv     = (const float*)d_in[8];
  const float* wo     = (const float*)d_in[9];
  const float* fg     = (const float*)d_in[10];
  const float* fb     = (const float*)d_in[11];
  const float* fw1    = (const float*)d_in[12];
  const float* fb1    = (const float*)d_in[13];
  const float* fw2    = (const float*)d_in[14];
  const float* fb2    = (const float*)d_in[15];
  const float* pn_g   = (const float*)d_in[16];
  const float* pn_b   = (const float*)d_in[17];
  const float* pwq    = (const float*)d_in[18];
  const float* pwk    = (const float*)d_in[19];
  const float* pwv    = (const float*)d_in[20];
  const float* pwo    = (const float*)d_in[21];
  const float* pfg    = (const float*)d_in[22];
  const float* pfb    = (const float*)d_in[23];
  const float* pfw1   = (const float*)d_in[24];
  const float* pfb1   = (const float*)d_in[25];
  const float* pfw2   = (const float*)d_in[26];
  const float* pfb2   = (const float*)d_in[27];
  const float* fuse_w = (const float*)d_in[28];
  const float* fuse_b = (const float*)d_in[29];
  const float* cls_w  = (const float*)d_in[30];
  const float* cls_b  = (const float*)d_in[31];

  const int NN = 128001, R = 128000, C = 512, INC = 256, OUTC = 40, Pn = 2000;

  size_t off = 0;
  auto alloc = [&](size_t elems) -> bf16* {
    bf16* p = (bf16*)((char*)d_ws + off);
    off += ((elems * sizeof(bf16)) + 255) & ~(size_t)255;
    return p;
  };

  bf16* encT = alloc((size_t)C * INC);
  bf16* clsT = alloc((size_t)OUTC * C);
  bf16 *wqT[2], *wkT[2], *wvT[2], *woT[2], *fw1T[2], *fw2T[2];
  bf16 *pwqT[2], *pwkT[2], *pwvT[2], *pwoT[2], *pfw1T[2], *pfw2T[2], *fuseT[2];
  for (int l = 0; l < 2; ++l) {
    wqT[l] = alloc((size_t)C * C);  wkT[l] = alloc((size_t)C * C);
    wvT[l] = alloc((size_t)C * C);  woT[l] = alloc((size_t)C * C);
    fw1T[l] = alloc((size_t)C * C); fw2T[l] = alloc((size_t)C * C);
    pwqT[l] = alloc((size_t)C * C); pwkT[l] = alloc((size_t)C * C);
    pwvT[l] = alloc((size_t)C * C); pwoT[l] = alloc((size_t)C * C);
    pfw1T[l] = alloc((size_t)C * C); pfw2T[l] = alloc((size_t)C * C);
    fuseT[l] = alloc((size_t)C * 2 * C);
  }
  bf16* Xb = alloc((size_t)NN * INC);
  bf16* Hbuf[2] = { alloc((size_t)NN * C), alloc((size_t)NN * C) };
  bf16* B0 = alloc((size_t)R * C);
  bf16* B1 = alloc((size_t)R * C);
  bf16* B2 = alloc((size_t)R * C);
  bf16* B3 = alloc((size_t)R * C);
  bf16* SM = alloc((size_t)Pn * C);
  bf16* S0 = alloc((size_t)Pn * C);
  bf16* S1 = alloc((size_t)Pn * C);
  bf16* S2 = alloc((size_t)Pn * C);
  bf16* S3 = alloc((size_t)Pn * C);

  auto cvtT = [&](const float* W, bf16* Wt, int K, int Ncols) {
    int tot = K * Ncols;
    cvtT_kernel<<<dim3((tot + 255) / 256), dim3(256), 0, stream>>>(W, Wt, K, Ncols);
  };
  cvtT(enc_w, encT, INC, C);
  cvtT(cls_w, clsT, C, OUTC);
  for (int l = 0; l < 2; ++l) {
    size_t wo_ = (size_t)l * C * C;
    cvtT(wq + wo_, wqT[l], C, C);   cvtT(wk + wo_, wkT[l], C, C);
    cvtT(wv + wo_, wvT[l], C, C);   cvtT(wo + wo_, woT[l], C, C);
    cvtT(fw1 + wo_, fw1T[l], C, C); cvtT(fw2 + wo_, fw2T[l], C, C);
    cvtT(pwq + wo_, pwqT[l], C, C); cvtT(pwk + wo_, pwkT[l], C, C);
    cvtT(pwv + wo_, pwvT[l], C, C); cvtT(pwo + wo_, pwoT[l], C, C);
    cvtT(pfw1 + wo_, pfw1T[l], C, C); cvtT(pfw2 + wo_, pfw2T[l], C, C);
    cvtT(fuse_w + (size_t)l * 2 * C * C, fuseT[l], 2 * C, C);
  }
  {
    long long n = (long long)NN * INC;
    cvt_kernel<<<dim3((unsigned)((n + 255) / 256)), dim3(256), 0, stream>>>(X, Xb, n);
  }

  auto gemm = [&](const bf16* A, int lda, const bf16* A2, const bf16* Bt,
                  const float* bias, const bf16* resid, bf16* outb, float* outf,
                  const int* scatter, int M, int Ncols, int K, int relu) {
    dim3 g((Ncols + 127) / 128, (M + 127) / 128);
    gemm_bf16_kernel<<<g, dim3(256), 0, stream>>>(A, lda, A2, Bt, bias, resid,
                                                  outb, outf, scatter, M, Ncols, K, relu);
  };
  auto ln = [&](const bf16* xin, bf16* yout, const float* g, const float* b, int M, float eps) {
    ln_kernel<<<dim3((M + 7) / 8), dim3(256), 0, stream>>>(xin, yout, g, b, M, eps);
  };

  // Attribute encoder: h = relu(x @ enc_w + enc_b)
  gemm(Xb, INC, nullptr, encT, enc_b, nullptr, Hbuf[0], nullptr, nullptr, NN, C, INC, 1);

  for (int l = 0; l < 2; ++l) {
    bf16* hin = Hbuf[l];
    bf16* hout = Hbuf[1 - l];  // holds hn; fuse scatters patch rows into it
    ln(hin, hout, nn_g + l * C, nn_b + l * C, NN, 1e-5f);
    gather_kernel<<<dim3(R), dim3(256), 0, stream>>>(hout, patch, B0, C);  // px
    gemm(B0, C, nullptr, wqT[l], nullptr, nullptr, B1, nullptr, nullptr, R, C, C, 0);
    gemm(B0, C, nullptr, wkT[l], nullptr, nullptr, B2, nullptr, nullptr, R, C, C, 0);
    gemm(B0, C, nullptr, wvT[l], nullptr, nullptr, B3, nullptr, nullptr, R, C, C, 0);
    node_attn_kernel<<<dim3(Pn, 8), dim3(128), 0, stream>>>(B1, B2, B3, B1, patch, C, NN - 1);
    gemm(B1, C, nullptr, woT[l], nullptr, B0, B2, nullptr, nullptr, R, C, C, 0);      // x1
    ln(B2, B3, fg + l * C, fb + l * C, R, 1e-6f);
    gemm(B3, C, nullptr, fw1T[l], fb1 + l * C, nullptr, B0, nullptr, nullptr, R, C, C, 1);
    gemm(B0, C, nullptr, fw2T[l], fb2 + l * C, B2, B1, nullptr, nullptr, R, C, C, 0); // x2
    meanpool_kernel<<<dim3((Pn * C + 255) / 256), dim3(256), 0, stream>>>(B1, SM, Pn, C);
    ln(SM, S0, pn_g + l * C, pn_b + l * C, Pn, 1e-5f);
    gemm(S0, C, nullptr, pwqT[l], nullptr, nullptr, S1, nullptr, nullptr, Pn, C, C, 0);
    gemm(S0, C, nullptr, pwkT[l], nullptr, nullptr, S2, nullptr, nullptr, Pn, C, C, 0);
    gemm(S0, C, nullptr, pwvT[l], nullptr, nullptr, S3, nullptr, nullptr, Pn, C, C, 0);
    patch_attn_kernel<<<dim3((Pn + 63) / 64, 8), dim3(128), 0, stream>>>(S1, S2, S3, S1, Pn, C);
    gemm(S1, C, nullptr, pwoT[l], nullptr, S0, SM, nullptr, nullptr, Pn, C, C, 0);    // x3
    ln(SM, S1, pfg + l * C, pfb + l * C, Pn, 1e-6f);
    gemm(S1, C, nullptr, pfw1T[l], pfb1 + l * C, nullptr, S2, nullptr, nullptr, Pn, C, C, 1);
    gemm(S2, C, nullptr, pfw2T[l], pfb2 + l * C, SM, S3, nullptr, nullptr, Pn, C, C, 0); // x4
    // fuse: relu([x2 | broadcast(x4)] @ fuse_w + fuse_b) + x2, scattered into hout
    gemm(B1, C, S3, fuseT[l], fuse_b + l * C, B1, hout, nullptr, patch, R, C, 2 * C, 1);
  }
  // Classifier: out = h @ cls_w + cls_b (fp32)
  gemm(Hbuf[0], C, nullptr, clsT, cls_b, nullptr, nullptr, (float*)d_out, nullptr, NN, OUTC, C, 0);
}
